// Controller_51084341019348
// MI455X (gfx1250) — matrix operand order repaired and checked by reading, not compiled, not run on hardware
//
#include <hip/hip_runtime.h>

// ---------------------------------------------------------------------------
// ENAS controller sampler for MI455X (gfx1250, wave32, WMMA bf16).
//   - 32 workgroups x 256 threads; each WG owns 16 rollouts (WMMA M=16).
//   - 8 waves/WG; wave w owns hidden-unit slice [w*256, w*256+256).
//   - LSTM gates via v_wmma_f32_16x16x32_bf16: A = x/h rows (LDS bf16),
//     B = weight rows (global bf16, converted once from fp32).
//   - h double-buffered in LDS (ping-pong); c / anchors / attn products in
//     per-WG global workspace slabs (per-node unique addresses + fences).
// ---------------------------------------------------------------------------

typedef float  v8f   __attribute__((ext_vector_type(8)));
typedef __bf16 v8bf  __attribute__((ext_vector_type(8)));
typedef __bf16 v16bf __attribute__((ext_vector_type(16)));

#define H_       2048
#define FOURH    8192
#define NNODE    14
#define BTOT     512
#define RPW      16                 // rollouts per workgroup (WMMA M)
#define NWG      (BTOT / RPW)       // 32 workgroups
#define NTHREADS 256                // 8 wave32s
#define OPS      8
#define TEMP     5.0f
#define TCONST   2.5f
#define EPSG     1e-10f

#define ADJ_ELEMS   (BTOT * NNODE * NNODE)   // 100352
#define OPS_ELEMS   (BTOT * 12)              // 6144
#define OFF_OPS     ADJ_ELEMS
#define OFF_LP      (ADJ_ELEMS + OPS_ELEMS)
#define OFF_ENT     (ADJ_ELEMS + OPS_ELEMS + BTOT)

__device__ __forceinline__ float sigmf(float x)   { return 1.f / (1.f + __expf(-x)); }
__device__ __forceinline__ float gumbelf(float u) { return -__logf(-__logf(u + EPSG) + EPSG); }

// A-fragment (16x32 bf16): lane = M row (lane%16); two 8-elem chunks at
// k + (lane/16)*8 and k + 16 + (lane/16)*8  (ISA 7.12.2 16-bit A layout).
__device__ __forceinline__ v16bf ldA(const __bf16* row, int kb, int hi) {
  union { v16bf v; v8bf h[2]; } u;
  u.h[0] = *(const v8bf*)(row + kb + hi * 8);
  u.h[1] = *(const v8bf*)(row + kb + 16 + hi * 8);
  return u.v;
}
// B-fragment (32x16 bf16): lane = column (lane%16); two 8-elem chunks at
// k + (lane/16)*8 and k + 16 + (lane/16)*8 of W row n (same K order as A).
__device__ __forceinline__ v16bf ldB(const __bf16* wrow, int kb, int hi) {
  union { v16bf v; v8bf h[2]; } u;
  u.h[0] = *(const v8bf*)(wrow + kb + hi * 8);
  u.h[1] = *(const v8bf*)(wrow + kb + 16 + hi * 8);
  return u.v;
}
__device__ __forceinline__ v8f wmma_bf(v16bf a, v16bf b, v8f c) {
  return __builtin_amdgcn_wmma_f32_16x16x32_bf16(false, a, false, b, (short)0, c, false, false);
}

// One batched LSTM step for this wave's 256-unit hidden slice.
// gates[m,n] for n in {g*2048 + hb + ht*16 + lane16}; C-tile element is
// (m = r + 8*hi, n = lane16) so the i/f/g/o combine is per-lane local.
__device__ void lstm_step(const __bf16* __restrict__ wih, const __bf16* __restrict__ whh,
                          const float* __restrict__ bih, const float* __restrict__ bhh,
                          float* __restrict__ cbuf,
                          const __bf16 (*xs)[H_], const __bf16 (*hold)[H_],
                          __bf16 (*hnew)[H_],
                          int wave, int lane16, int hi) {
  const int hb = wave * 256;
  const __bf16* xr = xs[lane16];
  const __bf16* hr = hold[lane16];
  for (int ht = 0; ht < 16; ++ht) {
    const int hu = hb + ht * 16 + lane16;
    const __bf16* wi0 = wih + (size_t)hu * H_;
    const __bf16* wi1 = wih + ((size_t)hu + 2048) * H_;
    const __bf16* wi2 = wih + ((size_t)hu + 4096) * H_;
    const __bf16* wi3 = wih + ((size_t)hu + 6144) * H_;
    const __bf16* wh0 = whh + (size_t)hu * H_;
    const __bf16* wh1 = whh + ((size_t)hu + 2048) * H_;
    const __bf16* wh2 = whh + ((size_t)hu + 4096) * H_;
    const __bf16* wh3 = whh + ((size_t)hu + 6144) * H_;
    if (ht < 15) {  // hint next tile's rows toward L1/L2 (global_prefetch_b8)
      __builtin_prefetch(wi0 + (size_t)16 * H_, 0, 1);
      __builtin_prefetch(wh0 + (size_t)16 * H_, 0, 1);
    }
    v8f a0 = {}, a1 = {}, a2 = {}, a3 = {};
#pragma unroll 2
    for (int kb = 0; kb < H_; kb += 32) {
      v16bf ax = ldA(xr, kb, hi);
      v16bf ah = ldA(hr, kb, hi);
      a0 = wmma_bf(ax, ldB(wi0, kb, hi), a0);
      a0 = wmma_bf(ah, ldB(wh0, kb, hi), a0);
      a1 = wmma_bf(ax, ldB(wi1, kb, hi), a1);
      a1 = wmma_bf(ah, ldB(wh1, kb, hi), a1);
      a2 = wmma_bf(ax, ldB(wi2, kb, hi), a2);
      a2 = wmma_bf(ah, ldB(wh2, kb, hi), a2);
      a3 = wmma_bf(ax, ldB(wi3, kb, hi), a3);
      a3 = wmma_bf(ah, ldB(wh3, kb, hi), a3);
    }
    const float bi0 = bih[hu]        + bhh[hu];
    const float bi1 = bih[hu + 2048] + bhh[hu + 2048];
    const float bi2 = bih[hu + 4096] + bhh[hu + 4096];
    const float bi3 = bih[hu + 6144] + bhh[hu + 6144];
#pragma unroll
    for (int r = 0; r < 8; ++r) {
      const int m = r + hi * 8;                   // rollout index
      const float iv = sigmf(a0[r] + bi0);
      const float fv = sigmf(a1[r] + bi1);
      const float gv = tanhf(a2[r] + bi2);
      const float ov = sigmf(a3[r] + bi3);
      const float cn = fv * cbuf[(size_t)m * H_ + hu] + iv * gv;
      cbuf[(size_t)m * H_ + hu] = cn;             // c owned by this wave only
      hnew[m][hu] = (__bf16)(ov * tanhf(cn));
    }
  }
  __syncthreads();   // publish hnew before anyone consumes it
}

// out[m, n] = sum_k h[m,k] * W[n,k] for this wave's 256-wide n slice (bf16 out).
__device__ void attn_mm(const __bf16* __restrict__ W, const __bf16 (*hs)[H_],
                        __bf16* __restrict__ outb, int wave, int lane16, int hi) {
  const int hb = wave * 256;
  const __bf16* hr = hs[lane16];
  for (int ht = 0; ht < 16; ++ht) {
    const int n = hb + ht * 16 + lane16;
    const __bf16* wr = W + (size_t)n * H_;
    v8f acc = {};
#pragma unroll 4
    for (int kb = 0; kb < H_; kb += 32)
      acc = wmma_bf(ldA(hr, kb, hi), ldB(wr, kb, hi), acc);
#pragma unroll
    for (int r = 0; r < 8; ++r)
      outb[(size_t)(r + hi * 8) * H_ + n] = (__bf16)acc[r];
  }
}

__global__ void cvt_bf16(const float* __restrict__ s, __bf16* __restrict__ d, int n) {
  int i = blockIdx.x * blockDim.x + threadIdx.x;
  if (i < n) d[i] = (__bf16)s[i];
}

__global__ void init_out(float* __restrict__ out, int n) {
  int i = blockIdx.x * blockDim.x + threadIdx.x;
  if (i < n) {
    float v = 0.f;
    if (i < ADJ_ELEMS && (i % (NNODE * NNODE)) == 1) v = 1.f;  // adj[b][0][1] = 1
    out[i] = v;
  }
}

__global__ __launch_bounds__(NTHREADS, 1)
void controller_main(const float* __restrict__ u_parent, const float* __restrict__ u_op,
                     const __bf16* __restrict__ emb_bf,
                     const __bf16* __restrict__ wih_bf, const float* __restrict__ b_ih,
                     const __bf16* __restrict__ whh_bf, const float* __restrict__ b_hh,
                     const float* __restrict__ w_soft, const float* __restrict__ b_soft,
                     const float* __restrict__ b_soft_nl,
                     const __bf16* __restrict__ w1_bf, const __bf16* __restrict__ w2_bf,
                     const float* __restrict__ v_attn,
                     float* __restrict__ cbuf_all, __bf16* __restrict__ w2h_all,
                     __bf16* __restrict__ anch_all, __bf16* __restrict__ aw1_all,
                     float* __restrict__ out) {
  const int tid = threadIdx.x;
  const int wave = tid >> 5, lane = tid & 31, lane16 = lane & 15, hi = lane >> 4;
  const int wg = blockIdx.x;

  float*  cbuf = cbuf_all + (size_t)wg * RPW * H_;
  __bf16* w2h  = w2h_all  + (size_t)wg * 12 * RPW * H_;      // per-node unique slabs
  __bf16* anch = anch_all + (size_t)wg * NNODE * RPW * H_;
  __bf16* aw1  = aw1_all  + (size_t)wg * NNODE * RPW * H_;

  __shared__ __bf16 x_s[RPW][H_];          // 64 KB
  __shared__ __bf16 h_s[2][RPW][H_];       // 128 KB ping-pong
  __shared__ float  red_s[RPW][16];
  __shared__ float  logit_s[RPW][16];
  __shared__ int    sel_s[RPW];

  // ---- init: c=0 (each wave zeroes its own slice), h=0, x=emb[0], anchors[0..1]=0
  for (int m = 0; m < RPW; ++m)
    for (int i = lane; i < 256; i += 32)
      cbuf[(size_t)m * H_ + wave * 256 + i] = 0.f;
  for (int i = tid; i < RPW * H_; i += NTHREADS) {
    ((__bf16*)h_s[0])[i] = (__bf16)0.f;
    ((__bf16*)x_s)[i]    = emb_bf[i & (H_ - 1)];
  }
  for (int i = tid; i < 2 * RPW * H_; i += NTHREADS) anch[i] = (__bf16)0.f;
  __threadfence();
  __syncthreads();

  int hcur = 0;
  // ---- warmup: 2 LSTM steps, anchors_w1[0..1] = w_attn_1 @ h
  for (int w2i = 0; w2i < 2; ++w2i) {
    lstm_step(wih_bf, whh_bf, b_ih, b_hh, cbuf, x_s, h_s[hcur], h_s[hcur ^ 1], wave, lane16, hi);
    hcur ^= 1;
    attn_mm(w1_bf, h_s[hcur], aw1 + (size_t)w2i * RPW * H_, wave, lane16, hi);
    __threadfence();
    __syncthreads();
  }

  float lp = 0.f, ent = 0.f;

  for (int lid = 2; lid < NNODE; ++lid) {
    // ===== step 1: parent attention + gumbel top-2 =====
    lstm_step(wih_bf, whh_bf, b_ih, b_hh, cbuf, x_s, h_s[hcur], h_s[hcur ^ 1], wave, lane16, hi);
    hcur ^= 1;
    __bf16* w2h_n = w2h + (size_t)(lid - 2) * RPW * H_;
    attn_mm(w2_bf, h_s[hcur], w2h_n, wave, lane16, hi);
    __threadfence();
    __syncthreads();

    {  // logits[b][j] = TC * tanh( v . tanh(aw1[j] + w2h) / T )
      const int b = tid >> 4, ch = tid & 15;
      const __bf16* wb = w2h_n + (size_t)b * H_;
      for (int j = 0; j < lid; ++j) {
        const __bf16* ab = aw1 + ((size_t)j * RPW + b) * H_;
        float s = 0.f;
        for (int k = ch * 128; k < ch * 128 + 128; ++k)
          s += v_attn[k] * tanhf((float)ab[k] + (float)wb[k]);
        red_s[b][ch] = s;
        __syncthreads();
        if (ch == 0) {
          float d = 0.f;
          for (int c2 = 0; c2 < 16; ++c2) d += red_s[b][c2];
          logit_s[b][j] = TCONST * tanhf(d / TEMP);
        }
        __syncthreads();
      }
    }
    if (tid < RPW) {  // one sampler thread per rollout
      const int b = tid, g = wg * RPW + b;
      float mx = -1e30f;
      for (int j = 0; j < lid; ++j) mx = fmaxf(mx, logit_s[b][j]);
      float se = 0.f;
      for (int j = 0; j < lid; ++j) se += __expf(logit_s[b][j] - mx);
      const float lse = mx + __logf(se);
      const float* up = u_parent + ((size_t)g * 12 + (lid - 2)) * NNODE;
      float p1 = -1e30f; int i0 = 0;
      for (int j = 0; j < lid; ++j) {
        float p = logit_s[b][j] + gumbelf(up[j]);
        if (p > p1) { p1 = p; i0 = j; }
      }
      float p2 = -1e30f; int i1 = 0;
      for (int j = 0; j < lid; ++j) {
        if (j == i0) continue;
        float p = logit_s[b][j] + gumbelf(up[j]);
        if (p > p2) { p2 = p; i1 = j; }
      }
      out[(size_t)g * (NNODE * NNODE) + i0 * NNODE + lid] = 1.f;
      out[(size_t)g * (NNODE * NNODE) + i1 * NNODE + lid] = 1.f;
      lp += 0.5f * (-(logit_s[b][i0] - lse) - (logit_s[b][i1] - lse));
      for (int j = 0; j < lid; ++j) {
        float l = logit_s[b][j] - lse;
        ent -= __expf(l) * l;
      }
      sel_s[b] = i1;  // idx[-1]
    }
    __syncthreads();
    {  // x = anchors[idx_last]
      const int m = tid >> 4, ch = tid & 15;
      const v8bf* src = (const v8bf*)(anch + ((size_t)sel_s[m] * RPW + m) * H_ + ch * 128);
      v8bf* dst = (v8bf*)(&x_s[m][ch * 128]);
#pragma unroll
      for (int k = 0; k < 16; ++k) dst[k] = src[k];
    }
    __syncthreads();

    // ===== step 2: op sampling =====
    lstm_step(wih_bf, whh_bf, b_ih, b_hh, cbuf, x_s, h_s[hcur], h_s[hcur ^ 1], wave, lane16, hi);
    hcur ^= 1;
    {  // w_soft @ h + b_soft -> 8 logits per rollout (tiny: VALU + LDS reduce)
      const int b = tid >> 4, q = tid & 15, o = q & 7, half = q >> 3;
      const float*  wr  = w_soft + (size_t)o * H_ + half * 1024;
      const __bf16* hb2 = &h_s[hcur][b][half * 1024];
      float s = 0.f;
      for (int k = 0; k < 1024; ++k) s += wr[k] * (float)hb2[k];
      red_s[b][q] = s;
      __syncthreads();
      if (q < OPS) {
        float v = red_s[b][q] + red_s[b][q + 8];
        logit_s[b][q] = tanhf((v + b_soft[q]) / TEMP) + b_soft_nl[q];  // TC/OTR == 1
      }
      __syncthreads();
    }
    if (tid < RPW) {
      const int b = tid, g = wg * RPW + b;
      float mx = -1e30f;
      for (int o = 0; o < OPS; ++o) mx = fmaxf(mx, logit_s[b][o]);
      float se = 0.f;
      for (int o = 0; o < OPS; ++o) se += __expf(logit_s[b][o] - mx);
      const float lse = mx + __logf(se);
      const float* uo = u_op + ((size_t)g * 12 + (lid - 2)) * OPS;
      float pb = -1e30f; int oi = 0;
      for (int o = 0; o < OPS; ++o) {
        float p = logit_s[b][o] + gumbelf(uo[o]);
        if (p > pb) { pb = p; oi = o; }
      }
      out[OFF_OPS + (size_t)g * 12 + (lid - 2)] = (float)oi;
      lp += -(logit_s[b][oi] - lse);
      for (int o = 0; o < OPS; ++o) {
        float l = logit_s[b][o] - lse;
        ent -= __expf(l) * l;
      }
      sel_s[b] = oi;
    }
    __syncthreads();
    {  // x = emb[op + 1]
      const int m = tid >> 4, ch = tid & 15;
      const v8bf* src = (const v8bf*)(emb_bf + (size_t)(sel_s[m] + 1) * H_ + ch * 128);
      v8bf* dst = (v8bf*)(&x_s[m][ch * 128]);
#pragma unroll
      for (int k = 0; k < 16; ++k) dst[k] = src[k];
    }
    __syncthreads();

    // ===== step 3: anchor update =====
    lstm_step(wih_bf, whh_bf, b_ih, b_hh, cbuf, x_s, h_s[hcur], h_s[hcur ^ 1], wave, lane16, hi);
    hcur ^= 1;
    {  // anchors[lid] = h; x = emb[0]
      const int m = tid >> 4, ch = tid & 15;
      const v8bf* hsrc = (const v8bf*)(&h_s[hcur][m][ch * 128]);
      v8bf* adst = (v8bf*)(anch + ((size_t)lid * RPW + m) * H_ + ch * 128);
      const v8bf* esrc = (const v8bf*)(emb_bf + ch * 128);
      v8bf* xdst = (v8bf*)(&x_s[m][ch * 128]);
#pragma unroll
      for (int k = 0; k < 16; ++k) { adst[k] = hsrc[k]; xdst[k] = esrc[k]; }
    }
    attn_mm(w1_bf, h_s[hcur], aw1 + (size_t)lid * RPW * H_, wave, lane16, hi);
    __threadfence();
    __syncthreads();
  }

  if (tid < RPW) {
    const int g = wg * RPW + tid;
    out[OFF_LP + g]  = lp;
    out[OFF_ENT + g] = ent;
  }
}

extern "C" void kernel_launch(void* const* d_in, const int* in_sizes, int n_in,
                              void* d_out, int out_size, void* d_ws, size_t ws_size,
                              hipStream_t stream) {
  const float* u_parent  = (const float*)d_in[0];
  const float* u_op      = (const float*)d_in[1];
  const float* emb       = (const float*)d_in[2];
  const float* w_ih      = (const float*)d_in[3];
  const float* b_ih      = (const float*)d_in[4];
  const float* w_hh      = (const float*)d_in[5];
  const float* b_hh      = (const float*)d_in[6];
  const float* w_soft    = (const float*)d_in[7];
  const float* b_soft    = (const float*)d_in[8];
  const float* b_soft_nl = (const float*)d_in[9];
  const float* w_attn_1  = (const float*)d_in[10];
  const float* w_attn_2  = (const float*)d_in[11];
  const float* v_attn    = (const float*)d_in[12];
  float* out = (float*)d_out;
  (void)in_sizes; (void)n_in; (void)ws_size;

  char* ws = (char*)d_ws;
  size_t off = 0;
  auto take = [&](size_t bytes) -> char* {
    char* p = ws + off;
    off += (bytes + 255) & ~(size_t)255;
    return p;
  };
  __bf16* wih_bf = (__bf16*)take((size_t)FOURH * H_ * 2);
  __bf16* whh_bf = (__bf16*)take((size_t)FOURH * H_ * 2);
  __bf16* w1_bf  = (__bf16*)take((size_t)H_ * H_ * 2);
  __bf16* w2_bf  = (__bf16*)take((size_t)H_ * H_ * 2);
  __bf16* emb_bf = (__bf16*)take((size_t)9 * H_ * 2);
  float*  cbuf   = (float*) take((size_t)NWG * RPW * H_ * 4);
  __bf16* w2hb   = (__bf16*)take((size_t)NWG * 12 * RPW * H_ * 2);
  __bf16* anch   = (__bf16*)take((size_t)NWG * NNODE * RPW * H_ * 2);
  __bf16* aw1    = (__bf16*)take((size_t)NWG * NNODE * RPW * H_ * 2);

  const int n1 = FOURH * H_;
  cvt_bf16<<<(n1 + 255) / 256, 256, 0, stream>>>(w_ih, wih_bf, n1);
  cvt_bf16<<<(n1 + 255) / 256, 256, 0, stream>>>(w_hh, whh_bf, n1);
  const int n2 = H_ * H_;
  cvt_bf16<<<(n2 + 255) / 256, 256, 0, stream>>>(w_attn_1, w1_bf, n2);
  cvt_bf16<<<(n2 + 255) / 256, 256, 0, stream>>>(w_attn_2, w2_bf, n2);
  const int n3 = 9 * H_;
  cvt_bf16<<<(n3 + 255) / 256, 256, 0, stream>>>(emb, emb_bf, n3);

  init_out<<<(out_size + 255) / 256, 256, 0, stream>>>(out, out_size);

  controller_main<<<NWG, NTHREADS, 0, stream>>>(
      u_parent, u_op, emb_bf, wih_bf, b_ih, whh_bf, b_hh,
      w_soft, b_soft, b_soft_nl, w1_bf, w2_bf, v_attn,
      cbuf, w2hb, anch, aw1, out);
}
